// CNN_22325240004626
// MI455X (gfx1250) — matrix operand order repaired and checked by reading, not compiled, not run on hardware
//
#include <hip/hip_runtime.h>

typedef __attribute__((ext_vector_type(16))) _Float16 v16h;
typedef __attribute__((ext_vector_type(8)))  _Float16 v8h;
typedef __attribute__((ext_vector_type(8)))  float    v8f;

#define B_    32
#define T_    8192
#define F_    16
#define HID_  128
#define OUT_  24
#define K1_   10
#define TB_   128          // timesteps per workgroup
#define HROWS 144          // h rows held in LDS (137 needed, padded to 9 tiles of 16)
#define HPAD  136          // h row stride in halfs (272B = 68 dwords -> bank-rotated, 16B aligned)
#define XROWS 160          // x rows held in LDS (153 needed)
#define XPAD  24           // x row stride in halfs (48B, 16B aligned, bank-rotated)

#define WB_ELEMS (5*8*32*16)     // 20480 halfs: conv1 B tiles [kc][nt][lane][e]
#define WF_ELEMS (10*4*2*32*16)  // 40960 halfs: conv2 B tiles [k][cc][nt][lane][e]

// ---------------------------------------------------------------------------
// Pack w1/wf into per-lane WMMA B-operand layout (f32 -> f16).
// B layout (16-bit, 32x16): col n = lane&15 ; K = (lane>=16 ? 8 : 0) + (e&7) + (e>=8 ? 16 : 0).
// ---------------------------------------------------------------------------
__global__ __launch_bounds__(256) void cnn_pack(const float* __restrict__ w1,
                                                const float* __restrict__ wf,
                                                _Float16* __restrict__ wb,
                                                _Float16* __restrict__ wfp)
{
    int i = blockIdx.x * 256 + threadIdx.x;
    if (i < WB_ELEMS) {
        int e    = i & 15;
        int lane = (i >> 4) & 31;
        int nt   = (i >> 9) & 7;
        int kc   = i >> 12;
        int K = kc * 32 + ((lane >= 16) ? 8 : 0) + (e & 7) + ((e & 8) << 1);   // 0..159
        int k = K >> 4;                                  // tap 0..9
        int f = K & 15;                                  // feature 0..15
        int n = nt * 16 + (lane & 15);                   // hid channel
        wb[i] = (_Float16)w1[(n * F_ + f) * K1_ + k];
    }
    if (i < WF_ELEMS) {
        int e    = i & 15;
        int lane = (i >> 4) & 31;
        int nt   = (i >> 9) & 1;
        int cc   = (i >> 10) & 3;
        int k    = i >> 12;
        int c = cc * 32 + ((lane >= 16) ? 8 : 0) + (e & 7) + ((e & 8) << 1);   // 0..127
        int o = nt * 16 + (lane & 15);                   // out channel (pad 24..31 with 0)
        wfp[i] = (o < OUT_) ? (_Float16)wf[o * (HID_ * K1_) + k * HID_ + c]
                            : (_Float16)0.0f;
    }
}

// ---------------------------------------------------------------------------
// Fused conv1 -> edge-replicate -> conv2. One WG = one batch x 128 timesteps.
// 256 threads = 8 wave32. VGPRs capped to 256 (4 waves/SIMD -> 2 WGs/WGP).
// ---------------------------------------------------------------------------
__global__ __launch_bounds__(256, 4) void cnn_fused(const float* __restrict__ x,
                                                    const float* __restrict__ b1,
                                                    const float* __restrict__ bf,
                                                    const _Float16* __restrict__ wb,
                                                    const _Float16* __restrict__ wfp,
                                                    float* __restrict__ out)
{
    __shared__ __align__(16) _Float16 xs[XROWS][XPAD];   //  7,680 B
    __shared__ __align__(16) _Float16 hs[HROWS][HPAD];   // 39,168 B

    const int wg   = blockIdx.x;
    const int b    = wg >> 6;             // 64 tiles per batch
    const int t0   = (wg & 63) * TB_;
    const int tid  = threadIdx.x;
    const int lane = tid & 31;
    const int wave = tid >> 5;
    const int lm   = lane & 15;
    const int half = lane >> 4;
    const int kbA  = half * 8;            // A-operand K base (ISA 16-bit A layout)

    // ---- stage x halo [t0-18, t0+141] into LDS as f16, zero-padded ----
    for (int idx = tid; idx < XROWS * F_; idx += 256) {
        int r = idx >> 4;
        int f = idx & 15;
        int t = t0 - 18 + r;
        float v = 0.0f;
        if (t >= 0 && t < T_) v = x[((size_t)b * T_ + t) * F_ + f];
        xs[r][f] = (_Float16)v;
    }
    __syncthreads();

    // ---- stage 1: h[r] (r=0..143 <-> t = t0-9+r) ----
    // 72 output units (9 row-tiles x 8 n-tiles), 9 per wave: balanced.
#pragma unroll 1
    for (int u = wave; u < 72; u += 8) {
        const int rt = u >> 3;
        const int nt = u & 7;
        v8f acc;
        {
            float bv = b1[nt * 16 + lm];
#pragma unroll
            for (int j = 0; j < 8; ++j) acc[j] = bv;
        }
        const int r = rt * 16 + lm;       // this lane's A row (h row index)
#pragma unroll
        for (int kc = 0; kc < 5; ++kc) {
            // A elements: two contiguous K-runs of 8 -> two 16B LDS loads
            int K0 = kc * 32 + kbA;       // run 0: K0..K0+7 (fixed tap k0)
            int k0 = K0 >> 4, f0 = K0 & 15;
            int Kb = K0 + 16;             // run 1: Kb..Kb+7 (fixed tap k1)
            int k1 = Kb >> 4, f1 = Kb & 15;
            union { v16h v; v8h h[2]; } a;
            a.h[0] = *(const v8h*)&xs[r + k0][f0];
            a.h[1] = *(const v8h*)&xs[r + k1][f1];
            v16h bm = *(const v16h*)(wb + ((((kc << 3) + nt) * 32 + lane) << 4));
            acc = __builtin_amdgcn_wmma_f32_16x16x32_f16(
                false, a.v, false, bm, (short)0, acc, false, false);
        }
        // D layout: VGPR rr -> row rr (lanes 0-15) / rr+8 (lanes 16-31), col = lane&15
#pragma unroll
        for (int rr = 0; rr < 8; ++rr) {
            hs[rt * 16 + rr + half * 8][nt * 16 + lm] = (_Float16)acc[rr];
        }
    }
    __syncthreads();

    // ---- edge replicate: first tile replaces h[t<0] slots with h[0] (row 9) ----
    if (t0 == 0) {
        for (int idx = tid; idx < 9 * HID_; idx += 256) {
            int rr = idx >> 7;
            int c  = idx & 127;
            hs[rr][c] = hs[9][c];
        }
    }
    __syncthreads();

    // ---- stage 2: y[t0+tb+m, o] = bf + sum_k wf_k . h_lds[tb+m+k] ----
    const int tb = wave * 16;
    v8f acc2[2];
#pragma unroll
    for (int nt = 0; nt < 2; ++nt) {
        int o = nt * 16 + lm;
        float bv = (o < OUT_) ? bf[o] : 0.0f;
#pragma unroll
        for (int j = 0; j < 8; ++j) acc2[nt][j] = bv;
    }
    const int arow = tb + lm;
#pragma unroll 1
    for (int k = 0; k < 10; ++k) {
#pragma unroll
        for (int cc = 0; cc < 4; ++cc) {
            union { v16h v; v8h h[2]; } a;
            int c0 = cc * 32 + kbA;
            a.h[0] = *(const v8h*)&hs[arow + k][c0];
            a.h[1] = *(const v8h*)&hs[arow + k][c0 + 16];
#pragma unroll
            for (int nt = 0; nt < 2; ++nt) {
                v16h bm = *(const v16h*)(wfp + (((((k << 2) + cc) * 2 + nt) * 32 + lane) << 4));
                acc2[nt] = __builtin_amdgcn_wmma_f32_16x16x32_f16(
                    false, a.v, false, bm, (short)0, acc2[nt], false, false);
            }
        }
    }

    // ---- epilogue: (B, T, OUT) f32, 16-lane-contiguous stores ----
#pragma unroll
    for (int nt = 0; nt < 2; ++nt) {
        int o = nt * 16 + lm;
        if (o < OUT_) {
#pragma unroll
            for (int rr = 0; rr < 8; ++rr) {
                int t = t0 + tb + rr + half * 8;
                out[((size_t)b * T_ + t) * OUT_ + o] = acc2[nt][rr];
            }
        }
    }
}

// ---------------------------------------------------------------------------
extern "C" void kernel_launch(void* const* d_in, const int* in_sizes, int n_in,
                              void* d_out, int out_size, void* d_ws, size_t ws_size,
                              hipStream_t stream)
{
    const float* x  = (const float*)d_in[0];
    const float* w1 = (const float*)d_in[1];
    const float* b1 = (const float*)d_in[2];
    const float* wf = (const float*)d_in[3];
    const float* bf = (const float*)d_in[4];

    _Float16* wb  = (_Float16*)d_ws;          // 40,960 B
    _Float16* wfp = wb + WB_ELEMS;            // 81,920 B  (total 122,880 B of ws)

    cnn_pack<<<(WF_ELEMS + 255) / 256, 256, 0, stream>>>(w1, wf, wb, wfp);

    const int n_wg = B_ * (T_ / TB_);         // 32 * 64 = 2048
    cnn_fused<<<n_wg, 256, 0, stream>>>(x, b1, bf, wb, wfp, (float*)d_out);
}
